// MVNet_28767690949501
// MI455X (gfx1250) — matrix operand order repaired and checked by reading, not compiled, not run on hardware
//
#include <hip/hip_runtime.h>
#include <hip/hip_bf16.h>

typedef __attribute__((ext_vector_type(16))) _Float16       v16h;
typedef __attribute__((ext_vector_type(8)))  float          v8f;
typedef __attribute__((ext_vector_type(8)))  unsigned short ushort8;

#define N_IMG   16
#define C_CH    256
#define HW      16384          // 128*128
#define P_TOT   (N_IMG * HW)   // 262144 pixels
#define NCLS    9
#define TILE    512            // pixels per workgroup tile (HW % TILE == 0)
#define THREADS 512            // 16 wave32 -> 16 channel groups of 16
#define KSTEPS  (TILE / 32)

// -------------------------------------------------------------------------
// Kernel 0: zero the workspace accumulators (d_ws is NOT re-zeroed between
// timed replays, so we must clear it every launch).
// -------------------------------------------------------------------------
__global__ void proto_zero(float* __restrict__ ws, int n) {
    int i = blockIdx.x * blockDim.x + threadIdx.x;
    if (i < n) ws[i] = 0.0f;
}

// -------------------------------------------------------------------------
// Kernel 1: fused argmax-label + one-hot GEMM segment reduction via WMMA.
//   sums:   [16 classes (9 used) x 256 channels] f32, atomically accumulated
//   counts: [16] f32 (9 used)
// -------------------------------------------------------------------------
__global__ __launch_bounds__(THREADS, 1)
void proto_main(const float* __restrict__ feat,
                const float* __restrict__ logit,
                float* __restrict__ sums,
                float* __restrict__ counts) {
    __shared__ alignas(16) unsigned short lab[TILE];
    __shared__ float cnt_s[NCLS];

    const int tid  = threadIdx.x;
    const int wave = tid >> 5;        // 0..15 -> channel group
    const int lane = tid & 31;
    const int m    = lane & 15;       // class row for A, channel col for D
    const int half = lane >> 4;       // A half select
    const int c0   = wave * 16;       // first channel of this wave's group

    float cnt[NCLS];
#pragma unroll
    for (int c = 0; c < NCLS; ++c) cnt[c] = 0.0f;

    v8f acc = {};                     // f32 16x16: M=class, N=channel

    for (int tile = blockIdx.x; tile < P_TOT / TILE; tile += gridDim.x) {
        const int pbase  = tile * TILE;
        const int n      = pbase / HW;        // tile lies within one image
        const int hwbase = pbase - n * HW;

        // ---- phase 1: labels (argmax over 9 logits, first-max wins) ----
        {
            const int i = tid;                // TILE == THREADS
            const float* lg = logit + (size_t)n * NCLS * HW + hwbase + i;
            float best = lg[0];
            int   bi   = 0;
#pragma unroll
            for (int c2 = 1; c2 < NCLS; ++c2) {
                float v = lg[(size_t)c2 * HW];
                if (v > best) { best = v; bi = c2; }
            }
            lab[i] = (unsigned short)bi;
#pragma unroll
            for (int c = 0; c < NCLS; ++c) cnt[c] += (bi == c) ? 1.0f : 0.0f;
        }
        __syncthreads();

        // ---- phase 2: one-hot WMMA accumulation over this tile ----
        const float* fb = feat + ((size_t)n * C_CH + c0) * HW + hwbase;
        for (int ks = 0; ks < KSTEPS; ++ks) {
            const int k0 = ks * 32;

            // A (16x32 f16 one-hot): halves 0..7 <- K = k0+8*half+e,
            //                        halves 8..15 <- K = k0+16+8*half+e
            ushort8 L1 = *(const ushort8*)&lab[k0 + 8 * half];
            ushort8 L2 = *(const ushort8*)&lab[k0 + 16 + 8 * half];
            v16h a;
#pragma unroll
            for (int e = 0; e < 8; ++e) {
                a[e]     = (L1[e] == (unsigned short)m) ? (_Float16)1.0f : (_Float16)0.0f;
                a[8 + e] = (L2[e] == (unsigned short)m) ? (_Float16)1.0f : (_Float16)0.0f;
            }

            // B (32x16 f16): lane%16 = channel, elements = pixel K in A's K order.
            // hi/lo split of f32 feat for ~f32-accurate accumulation.
            const float* fp = fb + (size_t)m * HW + k0;
            __builtin_prefetch(fp + 32, 0, 0);   // global_prefetch_b8 next step
            v16h bh, bl;
#pragma unroll
            for (int j = 0; j < 16; ++j) {
                float    x = fp[(j < 8) ? (8 * half + j) : (16 + 8 * half + (j - 8))];  // K = table order
                _Float16 h = (_Float16)x;
                bh[j] = h;
                bl[j] = (_Float16)(x - (float)h);
            }

            acc = __builtin_amdgcn_wmma_f32_16x16x32_f16(
                      false, a, false, bh, (short)0, acc, false, false);
            acc = __builtin_amdgcn_wmma_f32_16x16x32_f16(
                      false, a, false, bl, (short)0, acc, false, false);
        }
        __syncthreads();   // protect lab before next tile's phase 1
    }

    // ---- epilogue: fold private accumulators into global workspace ----
    if (tid < NCLS) cnt_s[tid] = 0.0f;
    __syncthreads();
#pragma unroll
    for (int c = 0; c < NCLS; ++c) atomicAdd(&cnt_s[c], cnt[c]);
    __syncthreads();
    if (tid < NCLS) atomicAdd(&counts[tid], cnt_s[tid]);

    // D layout: lane, reg r -> M = r + 8*(lane/16) (class), N = lane%16 (chan)
#pragma unroll
    for (int r = 0; r < 8; ++r) {
        const int M  = r + 8 * half;
        const int Nc = c0 + m;
        atomicAdd(&sums[M * C_CH + Nc], acc[r]);
    }
}

// -------------------------------------------------------------------------
// Kernel 2: protos = sums / max(count,1), broadcast over the batch dim.
// out[n][c][cls] = sums[cls][c] / max(counts[cls], 1)
// -------------------------------------------------------------------------
__global__ void proto_finalize(const float* __restrict__ sums,
                               const float* __restrict__ counts,
                               float* __restrict__ out) {
    int i = blockIdx.x * blockDim.x + threadIdx.x;
    if (i >= N_IMG * C_CH * NCLS) return;
    int cls = i % NCLS;
    int c   = (i / NCLS) % C_CH;
    out[i] = sums[cls * C_CH + c] / fmaxf(counts[cls], 1.0f);
}

// -------------------------------------------------------------------------
extern "C" void kernel_launch(void* const* d_in, const int* in_sizes, int n_in,
                              void* d_out, int out_size, void* d_ws, size_t ws_size,
                              hipStream_t stream) {
    const float* feat  = (const float*)d_in[0];
    const float* logit = (const float*)d_in[1];
    // d_in[2] = cls_num (scalar 9), hardcoded as NCLS.

    float* sums   = (float*)d_ws;           // 16*256 f32
    float* counts = sums + 16 * C_CH;       // 16 f32
    const int nws = 16 * C_CH + 16;

    proto_zero<<<(nws + 255) / 256, 256, 0, stream>>>(sums, nws);

    proto_main<<<P_TOT / TILE, THREADS, 0, stream>>>(feat, logit, sums, counts);

    const int nout = N_IMG * C_CH * NCLS;   // 36864
    proto_finalize<<<(nout + 255) / 256, 256, 0, stream>>>(sums, counts,
                                                           (float*)d_out);
}
